// AttentionMLPReduction_7945689497685
// MI455X (gfx1250) — matrix operand order repaired and checked by reading, not compiled, not run on hardware
//
#include <hip/hip_runtime.h>

// ---------------------------------------------------------------------------
// Types for CDNA5 WMMA (wave32, 16x16x32 bf16 -> f32)
// ---------------------------------------------------------------------------
typedef __attribute__((ext_vector_type(16))) __bf16 v16bf;
typedef __attribute__((ext_vector_type(8)))  __bf16 v8bf;
typedef __attribute__((ext_vector_type(4)))  __bf16 v4bf;
typedef __attribute__((ext_vector_type(8)))  float  v8f;

#define B_   256
#define L_   32
#define H_   4096
#define E_   2048
#define NH_  8
#define HD_  256
#define BL_  (B_ * L_)
#define QKV_LD (3 * E_)

static __device__ __forceinline__ __bf16 f2bf(float f) {
  unsigned u = __builtin_bit_cast(unsigned, f);
  u = (u + 0x7FFFu + ((u >> 16) & 1u)) >> 16;   // round-to-nearest-even
  return __builtin_bit_cast(__bf16, (unsigned short)u);
}

// A-fragment (16x32 bf16, row m striped by lane): lanes 0-15 hold row m with
// K = {k0..k0+7, k0+16..k0+23}; lanes 16-31 hold K = {k0+8..k0+15, k0+24..k0+31}.
static __device__ __forceinline__ v16bf load_a_frag(const __bf16* row, int k0, int lane) {
  const int base = k0 + ((lane & 16) ? 8 : 0);
  v8bf lo = *(const v8bf*)(row + base);
  v8bf hi = *(const v8bf*)(row + base + 16);
  v16bf r;
#pragma unroll
  for (int i = 0; i < 8; ++i) { r[i] = lo[i]; r[i + 8] = hi[i]; }
  return r;
}

// B-fragment (32x16 bf16) sourced from W stored N x K row-major (B = W^T):
// lane n holds column n; lanes 0-15: K = {k0..k0+7, k0+16..k0+23}, lanes 16-31: K = {k0+8..k0+15, k0+24..k0+31}.
static __device__ __forceinline__ v16bf load_b_frag(const __bf16* row, int k0, int lane) {
  const int base = k0 + ((lane & 16) ? 8 : 0);
  v8bf lo = *(const v8bf*)(row + base);
  v8bf hi = *(const v8bf*)(row + base + 16);
  v16bf r;
#pragma unroll
  for (int i = 0; i < 8; ++i) { r[i] = lo[i]; r[i + 8] = hi[i]; }
  return r;
}

#define WMMA_BF16(a, b, c) \
  __builtin_amdgcn_wmma_f32_16x16x32_bf16(false, (a), false, (b), (short)0, (c), false, false)

// ---------------------------------------------------------------------------
// fp32 -> bf16 staging (n4 = element count / 4)
// ---------------------------------------------------------------------------
__global__ __launch_bounds__(256)
void f32_to_bf16_kernel(const float* __restrict__ in, __bf16* __restrict__ out, int n4) {
  int i = blockIdx.x * 256 + threadIdx.x;
  if (i < n4) {
    const float4 v = ((const float4*)in)[i];
    v4bf o;
    o[0] = f2bf(v.x); o[1] = f2bf(v.y); o[2] = f2bf(v.z); o[3] = f2bf(v.w);
    ((v4bf*)out)[i] = o;
  }
}

// ---------------------------------------------------------------------------
// GEMM: out[m,n] = act( sum_k A[m,k] * W[n,k] + bias[n] )
// A: M x K bf16 row-major, W: N x K bf16 row-major.
// Block = 256 threads = 8 waves; wave tile 32(M) x 64(N); block tile 256 x 64.
// Per K-step: 4 A-frag loads + 8 B-frag loads feed 8 WMMAs (1.5 b128/WMMA).
// ---------------------------------------------------------------------------
template <bool RELU, bool BF16OUT>
__global__ __launch_bounds__(256)
void gemm_bt_kernel(const __bf16* __restrict__ A, const __bf16* __restrict__ W,
                    const float* __restrict__ bias, void* __restrict__ outp,
                    int M, int N, int K) {
  const int lane = threadIdx.x & 31;
  const int wv   = threadIdx.x >> 5;
  const int m0   = blockIdx.y * 256 + wv * 32;
  const int n0   = blockIdx.x * 64;

  const __bf16* arow0 = A + (size_t)(m0 + (lane & 15)) * K;
  const __bf16* arow1 = arow0 + (size_t)16 * K;
  const __bf16* brow  = W + (size_t)(n0 + (lane & 15)) * K;

  v8f acc[2][4] = {};
  for (int k0 = 0; k0 < K; k0 += 32) {
    const v16bf a0 = load_a_frag(arow0, k0, lane);
    const v16bf a1 = load_a_frag(arow1, k0, lane);
#pragma unroll
    for (int j = 0; j < 4; ++j) {
      const v16bf b = load_b_frag(brow + (size_t)(j * 16) * K, k0, lane);
      acc[0][j] = WMMA_BF16(a0, b, acc[0][j]);
      acc[1][j] = WMMA_BF16(a1, b, acc[1][j]);
    }
  }

  const int mh = (lane & 16) ? 8 : 0;
#pragma unroll
  for (int i = 0; i < 2; ++i) {
#pragma unroll
    for (int j = 0; j < 4; ++j) {
      const int n = n0 + j * 16 + (lane & 15);
      const float bv = bias[n];
#pragma unroll
      for (int r = 0; r < 8; ++r) {
        const int m = m0 + i * 16 + r + mh;
        float v = acc[i][j][r] + bv;
        if (RELU) v = fmaxf(v, 0.0f);
        if (BF16OUT) ((__bf16*)outp)[(size_t)m * N + n] = f2bf(v);
        else         ((float*)outp)[(size_t)m * N + n] = v;
      }
    }
  }
}

// ---------------------------------------------------------------------------
// Attention: one block per (b, head), 128 threads = 4 waves.
//   scores = q k^T / 16 (WMMA, 2x2 tiles, K=256)
//   softmax rows in LDS; colsum[b,h,l] = sum_q attn[q,l] (for pooling coeff)
//   ctx = attn @ v (WMMA, LDS-staged v^T and bf16 attn), stored bf16.
// ---------------------------------------------------------------------------
__global__ __launch_bounds__(128)
void attention_kernel(const __bf16* __restrict__ qkv, float* __restrict__ colsum,
                      __bf16* __restrict__ ctx) {
  const int h = blockIdx.x;          // head
  const int b = blockIdx.y;          // batch
  const int t = threadIdx.x;
  const int lane = t & 31;
  const int wv   = t >> 5;

  __shared__ float  s_sc[32][32];
  __shared__ __bf16 s_attn[32][32];
  __shared__ __bf16 s_vt[256][40];   // v^T with padded stride (16B-aligned rows)

  // ---- Phase 1: scores (each wave one 16x16 tile of the 32x32 block) ----
  {
    const int ti = wv >> 1, tj = wv & 1;
    const __bf16* qrow = qkv + (size_t)(b * L_ + ti * 16 + (lane & 15)) * QKV_LD + h * HD_;
    const __bf16* krow = qkv + (size_t)(b * L_ + tj * 16 + (lane & 15)) * QKV_LD + E_ + h * HD_;
    v8f acc = {};
    for (int k0 = 0; k0 < HD_; k0 += 32) {
      const v16bf a  = load_a_frag(qrow, k0, lane);
      const v16bf bb = load_b_frag(krow, k0, lane);
      acc = WMMA_BF16(a, bb, acc);
    }
    const int mh = (lane & 16) ? 8 : 0;
    const int n  = tj * 16 + (lane & 15);
#pragma unroll
    for (int r = 0; r < 8; ++r)
      s_sc[ti * 16 + r + mh][n] = acc[r] * 0.0625f;   // 1/sqrt(256)
  }
  __syncthreads();

  // ---- Phase 2a: row softmax ----
  if (t < 32) {
    float mx = -1e30f;
#pragma unroll
    for (int j = 0; j < 32; ++j) mx = fmaxf(mx, s_sc[t][j]);
    float sum = 0.0f;
#pragma unroll
    for (int j = 0; j < 32; ++j) {
      const float e = __expf(s_sc[t][j] - mx);
      s_sc[t][j] = e;
      sum += e;
    }
    const float inv = 1.0f / sum;
#pragma unroll
    for (int j = 0; j < 32; ++j) {
      const float v = s_sc[t][j] * inv;
      s_sc[t][j]   = v;
      s_attn[t][j] = f2bf(v);
    }
  }
  __syncthreads();

  // ---- Phase 2b: column sums for pooling coefficients ----
  if (t < 32) {
    float cs = 0.0f;
#pragma unroll
    for (int q = 0; q < 32; ++q) cs += s_sc[q][t];
    colsum[((size_t)b * NH_ + h) * L_ + t] = cs;
  }

  // ---- Phase 3: stage v^T into LDS ----
  for (int i = t; i < L_ * HD_; i += 128) {   // i = l*256 + d
    const int l = i >> 8, d = i & 255;
    s_vt[d][l] = qkv[(size_t)(b * L_ + l) * QKV_LD + 2 * E_ + h * HD_ + d];
  }
  __syncthreads();

  // ---- Phase 4: ctx = attn(32x32) @ v(32x256), single K-step per tile ----
  for (int tile = wv; tile < 32; tile += 4) {
    const int mt = tile & 1, nt = tile >> 1;
    const __bf16* arow = &s_attn[mt * 16 + (lane & 15)][0];
    const __bf16* brow = &s_vt[nt * 16 + (lane & 15)][0];
    const v16bf a  = load_a_frag(arow, 0, lane);
    const v16bf bb = load_b_frag(brow, 0, lane);
    v8f acc = {};
    acc = WMMA_BF16(a, bb, acc);
    const int mh = (lane & 16) ? 8 : 0;
    const int n  = nt * 16 + (lane & 15);
#pragma unroll
    for (int r = 0; r < 8; ++r) {
      const int m = mt * 16 + r + mh;
      ctx[(size_t)(b * L_ + m) * E_ + h * HD_ + n] = f2bf(acc[r]);
    }
  }
}

// ---------------------------------------------------------------------------
// coef[b,l] = (1/(L*NH)) * sum_h colsum[b,h,l]
// (softmax rows sum to 1, so w_norm == w_mean; pooling+query-mean collapses.)
// ---------------------------------------------------------------------------
__global__ __launch_bounds__(256)
void coef_kernel(const float* __restrict__ colsum, float* __restrict__ coef) {
  const int i = blockIdx.x * 256 + threadIdx.x;   // b*32 + l
  if (i < BL_) {
    const int b = i >> 5, l = i & 31;
    float s = 0.0f;
#pragma unroll
    for (int h = 0; h < NH_; ++h) s += colsum[((size_t)b * NH_ + h) * L_ + l];
    coef[i] = s * (1.0f / (L_ * NH_));
  }
}

// ---------------------------------------------------------------------------
// pooled[b,e] = sum_l coef[b,l] * attn_out[b,l,e]
// ---------------------------------------------------------------------------
__global__ __launch_bounds__(256)
void pool_kernel(const float* __restrict__ coef, const float* __restrict__ attn_out,
                 float* __restrict__ pooled) {
  const int b = blockIdx.y;
  const int e = blockIdx.x * 256 + threadIdx.x;
  const float* base = attn_out + (size_t)b * L_ * E_ + e;
  const float* c = coef + b * L_;
  float acc = 0.0f;
#pragma unroll
  for (int l = 0; l < L_; ++l) acc = fmaf(c[l], base[(size_t)l * E_], acc);
  pooled[(size_t)b * E_ + e] = acc;
}

// ---------------------------------------------------------------------------
// Fused MLP head: one block per batch row. w1..w4 stay L2-resident (~2.3 MB).
// ---------------------------------------------------------------------------
__global__ __launch_bounds__(256)
void mlp_head_kernel(const float* __restrict__ pooled,
                     const float* __restrict__ w1, const float* __restrict__ b1,
                     const float* __restrict__ w2, const float* __restrict__ b2,
                     const float* __restrict__ w3, const float* __restrict__ b3,
                     const float* __restrict__ w4, const float* __restrict__ b4,
                     float* __restrict__ out) {
  const int b = blockIdx.x;
  const int t = threadIdx.x;
  __shared__ float s_in[E_];
  __shared__ float s1[256];
  __shared__ float s2[128];
  __shared__ float s3[64];
  __shared__ float s4[64];

  for (int i = t; i < E_; i += 256) s_in[i] = pooled[(size_t)b * E_ + i];
  __syncthreads();

  {
    const float* wr = w1 + (size_t)t * E_;
    float acc = b1[t];
    for (int k = 0; k < E_; ++k) acc = fmaf(wr[k], s_in[k], acc);
    s1[t] = fmaxf(acc, 0.0f);
  }
  __syncthreads();
  if (t < 128) {
    const float* wr = w2 + (size_t)t * 256;
    float acc = b2[t];
#pragma unroll 4
    for (int k = 0; k < 256; ++k) acc = fmaf(wr[k], s1[k], acc);
    s2[t] = fmaxf(acc, 0.0f);
  }
  __syncthreads();
  if (t < 64) {
    const float* wr = w3 + (size_t)t * 128;
    float acc = b3[t];
#pragma unroll 4
    for (int k = 0; k < 128; ++k) acc = fmaf(wr[k], s2[k], acc);
    s3[t] = fmaxf(acc, 0.0f);
    s4[t] = s3[t] * w4[t];
  }
  __syncthreads();
  if (t == 0) {
    float acc = b4[0];
#pragma unroll
    for (int k = 0; k < 64; ++k) acc += s4[k];
    out[b] = 1.0f / (1.0f + __expf(-acc));
  }
}

// ---------------------------------------------------------------------------
// Host launcher
// ---------------------------------------------------------------------------
extern "C" void kernel_launch(void* const* d_in, const int* in_sizes, int n_in,
                              void* d_out, int out_size, void* d_ws, size_t ws_size,
                              hipStream_t stream) {
  (void)in_sizes; (void)n_in; (void)out_size; (void)ws_size;

  const float* x     = (const float*)d_in[0];
  const float* w_red = (const float*)d_in[1];
  const float* b_red = (const float*)d_in[2];
  const float* w_in  = (const float*)d_in[3];
  const float* b_in  = (const float*)d_in[4];
  const float* w_out = (const float*)d_in[5];
  const float* b_out = (const float*)d_in[6];
  const float* w1 = (const float*)d_in[7];   const float* b1 = (const float*)d_in[8];
  const float* w2 = (const float*)d_in[9];   const float* b2 = (const float*)d_in[10];
  const float* w3 = (const float*)d_in[11];  const float* b3 = (const float*)d_in[12];
  const float* w4 = (const float*)d_in[13];  const float* b4 = (const float*)d_in[14];
  float* out = (float*)d_out;

  char* ws = (char*)d_ws;
  size_t off = 0;
  auto alloc = [&](size_t bytes) -> char* {
    char* p = ws + off;
    off += (bytes + 255) & ~(size_t)255;
    return p;
  };

  __bf16* xb       = (__bf16*)alloc((size_t)BL_ * H_ * 2);        // 64 MB
  __bf16* wredb    = (__bf16*)alloc((size_t)E_ * H_ * 2);         // 16 MB
  __bf16* winb     = (__bf16*)alloc((size_t)3 * E_ * E_ * 2);     // 24 MB
  __bf16* woutb    = (__bf16*)alloc((size_t)E_ * E_ * 2);         //  8 MB
  __bf16* h_bf     = (__bf16*)alloc((size_t)BL_ * E_ * 2);        // 32 MB
  __bf16* qkv_bf   = (__bf16*)alloc((size_t)BL_ * 3 * E_ * 2);    // 96 MB
  __bf16* ctx_bf   = (__bf16*)alloc((size_t)BL_ * E_ * 2);        // 32 MB
  float*  attn_out = (float*)alloc((size_t)BL_ * E_ * 4);         // 64 MB
  float*  colsum   = (float*)alloc((size_t)B_ * NH_ * L_ * 4);
  float*  coef     = (float*)alloc((size_t)BL_ * 4);
  float*  pooled   = (float*)alloc((size_t)B_ * E_ * 4);

  // 1) Stage inputs/weights to bf16
  {
    const int nx = BL_ * H_ / 4, nr = E_ * H_ / 4, ni = 3 * E_ * E_ / 4, no = E_ * E_ / 4;
    f32_to_bf16_kernel<<<(nx + 255) / 256, 256, 0, stream>>>(x,     xb,    nx);
    f32_to_bf16_kernel<<<(nr + 255) / 256, 256, 0, stream>>>(w_red, wredb, nr);
    f32_to_bf16_kernel<<<(ni + 255) / 256, 256, 0, stream>>>(w_in,  winb,  ni);
    f32_to_bf16_kernel<<<(no + 255) / 256, 256, 0, stream>>>(w_out, woutb, no);
  }

  // 2) h = relu(x @ w_red^T + b_red)   (M=8192, N=2048, K=4096), bf16 out
  gemm_bt_kernel<true, true><<<dim3(E_ / 64, BL_ / 256), 256, 0, stream>>>(
      xb, wredb, b_red, h_bf, BL_, E_, H_);

  // 3) qkv = h @ w_in^T + b_in         (M=8192, N=6144, K=2048), bf16 out
  gemm_bt_kernel<false, true><<<dim3(3 * E_ / 64, BL_ / 256), 256, 0, stream>>>(
      h_bf, winb, b_in, qkv_bf, BL_, 3 * E_, E_);

  // 4) per-(b,head) attention -> ctx (bf16) + attn column sums
  attention_kernel<<<dim3(NH_, B_), 128, 0, stream>>>(qkv_bf, colsum, ctx_bf);

  // 5) attn_out = ctx @ w_out^T + b_out (M=8192, N=2048, K=2048), fp32 out
  gemm_bt_kernel<false, false><<<dim3(E_ / 64, BL_ / 256), 256, 0, stream>>>(
      ctx_bf, woutb, b_out, attn_out, BL_, E_, E_);

  // 6) pooling coefficients, 7) weighted pool
  coef_kernel<<<BL_ / 256, 256, 0, stream>>>(colsum, coef);
  pool_kernel<<<dim3(E_ / 256, B_), 256, 0, stream>>>(coef, attn_out, pooled);

  // 8) fused MLP head -> sigmoid
  mlp_head_kernel<<<B_, 256, 0, stream>>>(pooled, w1, b1, w2, b2, w3, b3, w4, b4, out);
}
